// TFMamba_34883724378766
// MI455X (gfx1250) — compile-verified, matrix operand order checked by reading, not run on hardware
//
#include <hip/hip_runtime.h>

// ---------------- model dims ----------------
#define BB   128
#define DD   192
#define DI_  384
#define DS_  16
#define DR_  12
#define LL   17
#define DEPTH_ 12
#define NH_  8
#define HD_  24
#define ML   (BB*LL)     // 2176 rows in sequence GEMMs
#define DBLP 48          // padded row stride for dbl (44 -> 48 for 16B alignment)
#define MT   4           // 4 x 16 = 64 output rows per wave

typedef __attribute__((ext_vector_type(16))) _Float16 v16h;
typedef __attribute__((ext_vector_type(8)))  float    v8f;

// Load one 16-element f16 fragment (this lane's half of a 16x32 tile row):
// 8 contiguous floats at p, 8 more at p+16.  p must be 16B aligned.
__device__ __forceinline__ v16h load_frag(const float* __restrict__ p)
{
    float4 x0 = *(const float4*)(p);
    float4 x1 = *(const float4*)(p + 4);
    float4 x2 = *(const float4*)(p + 16);
    float4 x3 = *(const float4*)(p + 20);
    v16h f;
    f[0]  = (_Float16)x0.x; f[1]  = (_Float16)x0.y;
    f[2]  = (_Float16)x0.z; f[3]  = (_Float16)x0.w;
    f[4]  = (_Float16)x1.x; f[5]  = (_Float16)x1.y;
    f[6]  = (_Float16)x1.z; f[7]  = (_Float16)x1.w;
    f[8]  = (_Float16)x2.x; f[9]  = (_Float16)x2.y;
    f[10] = (_Float16)x2.z; f[11] = (_Float16)x2.w;
    f[12] = (_Float16)x3.x; f[13] = (_Float16)x3.y;
    f[14] = (_Float16)x3.z; f[15] = (_Float16)x3.w;
    return f;
}

#define WMMA_F16(a, b, c) \
    __builtin_amdgcn_wmma_f32_16x16x32_f16(false, (a), false, (b), (short)0, (c), false, false)

// =====================================================================
// WMMA GEMM:  C[M,N] = act( A[M,K](lda) * W[N,K](ldw)^T + bias )
// Preconditions: M % 64 == 0, K % 32 == 0, lda/ldw multiples of 4,
// A/W 16B aligned.  N tail handled by W-row clamp + store guard.
// One wave computes a 64x16 strip: per K-step all 5 fragments (1 W + 4 A)
// are loaded into distinct registers first (clause-friendly), then 4
// back-to-back WMMAs.  Live set ~80 VGPRs -> no spills; latency hidden
// by multi-wave occupancy (32-thread workgroups).
// act: 0=none, 1=relu, 2=softplus
// =====================================================================
__global__ void k_gemm_wmma(const float* __restrict__ A, int lda,
                            const float* __restrict__ W, int ldw,
                            const float* __restrict__ bias,
                            float* __restrict__ C, int ldc,
                            int M, int N, int K, int act)
{
    const int lane  = threadIdx.x;       // 0..31
    const int n0    = blockIdx.x * 16;
    const int m0    = blockIdx.y * (16 * MT);
    const int half  = lane >> 4;         // K-half selector
    const int idx16 = lane & 15;

    const int wn = n0 + idx16;
    const float* __restrict__ Wrow = W + (size_t)(wn < N ? wn : 0) * ldw;
    const float* __restrict__ Arow = A + (size_t)(m0 + idx16) * lda;
    const size_t astep = (size_t)16 * lda;
    const int koff = half * 8;

    v8f acc0 = (v8f){0.f,0.f,0.f,0.f,0.f,0.f,0.f,0.f};
    v8f acc1 = acc0, acc2 = acc0, acc3 = acc0;

    for (int kb = 0; kb < K; kb += 32) {
        const int k0 = kb + koff;
        v16h bf  = load_frag(Wrow + k0);
        v16h af0 = load_frag(Arow + 0 * astep + k0);
        v16h af1 = load_frag(Arow + 1 * astep + k0);
        v16h af2 = load_frag(Arow + 2 * astep + k0);
        v16h af3 = load_frag(Arow + 3 * astep + k0);
        acc0 = WMMA_F16(af0, bf, acc0);
        acc1 = WMMA_F16(af1, bf, acc1);
        acc2 = WMMA_F16(af2, bf, acc2);
        acc3 = WMMA_F16(af3, bf, acc3);
    }

    const int n = n0 + idx16;
    if (n < N) {
        const float bv = bias ? bias[n] : 0.f;
        float outv[MT * 8];
        #pragma unroll
        for (int r = 0; r < 8; ++r) {
            outv[0 * 8 + r] = acc0[r];
            outv[1 * 8 + r] = acc1[r];
            outv[2 * 8 + r] = acc2[r];
            outv[3 * 8 + r] = acc3[r];
        }
        #pragma unroll
        for (int t = 0; t < MT; ++t) {
            #pragma unroll
            for (int r = 0; r < 8; ++r) {
                int m = m0 + t * 16 + r + half * 8;
                float v = outv[t * 8 + r] + bv;
                if (act == 1) v = fmaxf(v, 0.f);
                else if (act == 2) v = (v > 20.f) ? v : __logf(1.f + __expf(v));
                C[(size_t)m * ldc + n] = v;
            }
        }
    }
}

// zero-pad dt_w (DEPTH, DI, 12) -> (DEPTH, DI, 32)
__global__ void k_pad_dtw(const float* __restrict__ src, float* __restrict__ dst)
{
    int idx = blockIdx.x * blockDim.x + threadIdx.x;
    if (idx >= DEPTH_ * DI_ * 32) return;
    int k  = idx & 31;
    int rc = idx >> 5;                   // layer*DI + c
    dst[idx] = (k < DR_) ? src[rc * DR_ + k] : 0.f;
}

// =====================================================================
// im2col for the two patch embeddings (x: [128,8,224,16] row-major)
// =====================================================================
__global__ void k_im2col_t(const float* __restrict__ x, float* __restrict__ a)
{
    int idx = blockIdx.x * blockDim.x + threadIdx.x;
    const int total = 2048 * 1792;
    if (idx >= total) return;
    int k = idx % 1792;
    int m = idx / 1792;
    int h = m % 16, b = m / 16;
    int c = k / 224, rem = k % 224;      // rem = r*16 + w
    a[idx] = x[(size_t)(b * 8 + c) * 3584 + h * 224 + rem];
}

__global__ void k_im2col_f(const float* __restrict__ x, float* __restrict__ a)
{
    int idx = blockIdx.x * blockDim.x + threadIdx.x;
    const int total = 2048 * 1792;
    if (idx >= total) return;
    int k = idx % 1792;
    int m = idx / 1792;
    int w = m % 16, b = m / 16;
    int c = k / 224, r = k % 224;
    a[idx] = x[(size_t)(b * 8 + c) * 3584 + r * 16 + w];
}

// seq[b,l,d] = (l<16 ? embed[b*16+l,d] : tok[d]) + pos[l,d]
__global__ void k_assemble(const float* __restrict__ e, const float* __restrict__ tok,
                           const float* __restrict__ pos, float* __restrict__ seq)
{
    int idx = blockIdx.x * blockDim.x + threadIdx.x;
    if (idx >= BB * LL * DD) return;
    int d = idx % DD;
    int bl = idx / DD;
    int l = bl % LL, b = bl / LL;
    float v = (l < 16) ? e[(size_t)(b * 16 + l) * DD + d] : tok[d];
    seq[idx] = v + pos[l * DD + d];
}

// =====================================================================
// residual update + RMSNorm. One block (64 thr) per row of 192.
// =====================================================================
__global__ void k_resnorm(const float* __restrict__ x, float* __restrict__ res,
                          const float* __restrict__ nw, float* __restrict__ h,
                          int first)
{
    const int row = blockIdx.x;
    const int tid = threadIdx.x;
    __shared__ float red[64];
    const float* xr = x + (size_t)row * DD;
    float* rr = res + (size_t)row * DD;
    float vals[3];
    float ss = 0.f;
    #pragma unroll
    for (int i = 0; i < 3; ++i) {
        int d = tid + i * 64;
        float v = xr[d];
        if (!first) v += rr[d];
        vals[i] = v;
        ss += v * v;
    }
    red[tid] = ss;
    __syncthreads();
    for (int s = 32; s > 0; s >>= 1) {
        if (tid < s) red[tid] += red[tid + s];
        __syncthreads();
    }
    float r = rsqrtf(red[0] / (float)DD + 1e-5f);
    #pragma unroll
    for (int i = 0; i < 3; ++i) {
        int d = tid + i * 64;
        rr[d] = vals[i];
        h[(size_t)row * DD + d] = vals[i] * r * nw[d];
    }
}

// causal depthwise conv (k=4) over L, + bias, + SiLU.  input = xz[...,0:384]
__global__ void k_conv_silu(const float* __restrict__ xz, const float* __restrict__ cw,
                            const float* __restrict__ cb, float* __restrict__ xx)
{
    int idx = blockIdx.x * blockDim.x + threadIdx.x;
    if (idx >= BB * LL * DI_) return;
    int c = idx % DI_;
    int bl = idx / DI_;
    int l = bl % LL, b = bl / LL;
    float s = cb[c];
    #pragma unroll
    for (int k = 0; k < 4; ++k) {
        int ls = l - 3 + k;
        if (ls >= 0) s += xz[(size_t)(b * LL + ls) * (2 * DI_) + c] * cw[c * 4 + k];
    }
    xx[idx] = s / (1.f + __expf(-s));    // silu
}

// =====================================================================
// Selective-scan: one thread per (b, channel). 16-wide state in regs.
// dt already softplus'ed (fused into the dt GEMM epilogue).
// =====================================================================
__global__ void k_scan(const float* __restrict__ dt, const float* __restrict__ xx,
                       const float* __restrict__ dbl, const float* __restrict__ xz,
                       const float* __restrict__ A_log, const float* __restrict__ Dp,
                       float* __restrict__ y)
{
    int idx = blockIdx.x * blockDim.x + threadIdx.x;
    if (idx >= BB * DI_) return;
    int c = idx % DI_, b = idx / DI_;
    float Ac[DS_];
    #pragma unroll
    for (int n = 0; n < DS_; ++n) Ac[n] = -__expf(A_log[c * DS_ + n]);
    const float dp = Dp[c];
    float h[DS_];
    #pragma unroll
    for (int n = 0; n < DS_; ++n) h[n] = 0.f;

    for (int l = 0; l < LL; ++l) {
        size_t r = (size_t)(b * LL + l);
        float dtv = dt[r * DI_ + c];
        float xv  = xx[r * DI_ + c];
        const float* dr = dbl + r * DBLP;            // [dt(12) | B(16) | C(16) | pad]
        float acc = 0.f;
        #pragma unroll
        for (int n = 0; n < DS_; ++n) {
            float dA = __expf(dtv * Ac[n]);
            h[n] = dA * h[n] + dtv * dr[DR_ + n] * xv;
            acc += h[n] * dr[DR_ + DS_ + n];
        }
        float zv = xz[r * (2 * DI_) + DI_ + c];
        y[r * DI_ + c] = (acc + dp * xv) * (zv / (1.f + __expf(-zv)));
    }
}

// =====================================================================
// 17x17 cross-attention, head dim 24. One block (32 thr) per (b, head).
// =====================================================================
__global__ void k_attn(const float* __restrict__ q, const float* __restrict__ k,
                       const float* __restrict__ v, float* __restrict__ o)
{
    int bh = blockIdx.x;
    int head = bh % NH_, b = bh / NH_;
    __shared__ float ks[LL * HD_], vs[LL * HD_];
    int tid = threadIdx.x;
    for (int i = tid; i < LL * HD_; i += 32) {
        int l = i / HD_, d = i % HD_;
        size_t off = (size_t)(b * LL + l) * DD + head * HD_ + d;
        ks[i] = k[off];
        vs[i] = v[off];
    }
    __syncthreads();
    if (tid < LL) {
        float qv[HD_];
        size_t qoff = (size_t)(b * LL + tid) * DD + head * HD_;
        #pragma unroll
        for (int d = 0; d < HD_; ++d) qv[d] = q[qoff + d];
        float s[LL];
        float mx = -1e30f;
        for (int kk = 0; kk < LL; ++kk) {
            float dot = 0.f;
            #pragma unroll
            for (int d = 0; d < HD_; ++d) dot += qv[d] * ks[kk * HD_ + d];
            s[kk] = dot * 0.2041241452319315f;   // 1/sqrt(24)
            mx = fmaxf(mx, s[kk]);
        }
        float sum = 0.f;
        for (int kk = 0; kk < LL; ++kk) { s[kk] = __expf(s[kk] - mx); sum += s[kk]; }
        float inv = 1.f / sum;
        float ov[HD_];
        #pragma unroll
        for (int d = 0; d < HD_; ++d) ov[d] = 0.f;
        for (int kk = 0; kk < LL; ++kk) {
            float p = s[kk] * inv;
            #pragma unroll
            for (int d = 0; d < HD_; ++d) ov[d] += p * vs[kk * HD_ + d];
        }
        #pragma unroll
        for (int d = 0; d < HD_; ++d) o[qoff + d] = ov[d];
    }
}

// feat[b,d] = t[b, L-1, d] + fusion[b, L-1, d]
__global__ void k_feat(const float* __restrict__ t, const float* __restrict__ fus,
                       float* __restrict__ feat)
{
    int idx = blockIdx.x * blockDim.x + threadIdx.x;
    if (idx >= BB * DD) return;
    int d = idx % DD, b = idx / DD;
    size_t off = (size_t)(b * LL + (LL - 1)) * DD + d;
    feat[idx] = t[off] + fus[off];
}

// ---------------------------------------------------------------------
static inline void gemm(hipStream_t s, const float* A, int lda,
                        const float* W, int ldw, const float* bias,
                        float* C, int ldc, int M, int N, int K, int act)
{
    dim3 g((N + 15) / 16, M / (16 * MT));   // M % 64 == 0 for all calls
    k_gemm_wmma<<<g, 32, 0, s>>>(A, lda, W, ldw, bias, C, ldc, M, N, K, act);
}

static void run_backbone(hipStream_t s, float* seq, float* resb, float* hb,
                         float* xzb, float* xxb, float* dblb, float* dt_act, float* yb,
                         const float* nw, const float* inp, const float* cw,
                         const float* cb, const float* xp, const float* dtw_pad,
                         const float* dtbias, const float* alog, const float* dd,
                         const float* op)
{
    for (int i = 0; i < DEPTH_; ++i) {
        k_resnorm<<<ML, 64, 0, s>>>(seq, resb, nw + (size_t)i * DD, hb, i == 0 ? 1 : 0);
        gemm(s, hb, DD, inp + (size_t)i * 2 * DI_ * DD, DD, nullptr,
             xzb, 2 * DI_, ML, 2 * DI_, DD, 0);
        k_conv_silu<<<(ML * DI_ + 255) / 256, 256, 0, s>>>(
            xzb, cw + (size_t)i * DI_ * 4, cb + (size_t)i * DI_, xxb);
        gemm(s, xxb, DI_, xp + (size_t)i * (DR_ + 2 * DS_) * DI_, DI_, nullptr,
             dblb, DBLP, ML, DR_ + 2 * DS_, DI_, 0);
        gemm(s, dblb, DBLP, dtw_pad + (size_t)i * DI_ * 32, 32,
             dtbias + (size_t)i * DI_, dt_act, DI_, ML, DI_, 32, 2 /*softplus*/);
        k_scan<<<(BB * DI_ + 255) / 256, 256, 0, s>>>(
            dt_act, xxb, dblb, xzb, alog + (size_t)i * DI_ * DS_, dd + (size_t)i * DI_, yb);
        gemm(s, yb, DI_, op + (size_t)i * DD * DI_, DI_, nullptr, seq, DD, ML, DD, DI_, 0);
    }
}

static void run_xattn(hipStream_t s, const float* q, const float* kv,
                      const float* w, const float* b,
                      float* qb, float* kb, float* vb, float* ob, float* out)
{
    gemm(s, q,  DD, w + 0 * DD * DD, DD, b + 0 * DD, qb, DD, ML, DD, DD, 0);
    gemm(s, kv, DD, w + 1 * DD * DD, DD, b + 1 * DD, kb, DD, ML, DD, DD, 0);
    gemm(s, kv, DD, w + 2 * DD * DD, DD, b + 2 * DD, vb, DD, ML, DD, DD, 0);
    k_attn<<<BB * NH_, 32, 0, s>>>(qb, kb, vb, ob);
    gemm(s, ob, DD, w + 3 * DD * DD, DD, b + 3 * DD, out, DD, ML, DD, DD, 0);
}

extern "C" void kernel_launch(void* const* d_in, const int* in_sizes, int n_in,
                              void* d_out, int out_size, void* d_ws, size_t ws_size,
                              hipStream_t stream)
{
    (void)in_sizes; (void)n_in; (void)out_size; (void)ws_size;

    const float* X      = (const float*)d_in[0];
    const float* t_pw   = (const float*)d_in[1];
    const float* t_pb   = (const float*)d_in[2];
    const float* f_pw   = (const float*)d_in[3];
    const float* f_pb   = (const float*)d_in[4];
    const float* t_tok  = (const float*)d_in[5];
    const float* f_tok  = (const float*)d_in[6];
    const float* t_pos  = (const float*)d_in[7];
    const float* f_pos  = (const float*)d_in[8];
    const float* t_nw   = (const float*)d_in[9];
    const float* t_inp  = (const float*)d_in[10];
    const float* t_cw   = (const float*)d_in[11];
    const float* t_cb   = (const float*)d_in[12];
    const float* t_xp   = (const float*)d_in[13];
    const float* t_dtw  = (const float*)d_in[14];
    const float* t_dtbv = (const float*)d_in[15];
    const float* t_alog = (const float*)d_in[16];
    const float* t_dd   = (const float*)d_in[17];
    const float* t_op   = (const float*)d_in[18];
    const float* f_nw   = (const float*)d_in[19];
    const float* f_inp  = (const float*)d_in[20];
    const float* f_cw   = (const float*)d_in[21];
    const float* f_cb   = (const float*)d_in[22];
    const float* f_xp   = (const float*)d_in[23];
    const float* f_dtw  = (const float*)d_in[24];
    const float* f_dtbv = (const float*)d_in[25];
    const float* f_alog = (const float*)d_in[26];
    const float* f_dd   = (const float*)d_in[27];
    const float* f_op   = (const float*)d_in[28];
    const float* att_t_w  = (const float*)d_in[29];
    const float* att_t_b  = (const float*)d_in[30];
    const float* att_tf_w = (const float*)d_in[31];
    const float* att_tf_b = (const float*)d_in[32];
    const float* cls1_w = (const float*)d_in[33];
    const float* cls1_b = (const float*)d_in[34];
    const float* cls2_w = (const float*)d_in[35];
    const float* cls2_b = (const float*)d_in[36];
    const float* pos1_w = (const float*)d_in[37];
    const float* pos1_b = (const float*)d_in[38];
    const float* pos2_w = (const float*)d_in[39];
    const float* pos2_b = (const float*)d_in[40];

    float* ws  = (float*)d_ws;
    float* out = (float*)d_out;

    // ---- workspace layout (floats, all offsets 64B-aligned) ----
    // Region0: im2col (3,670,016) + patch-embed out (393,216); dead after
    // the backbones, so the 6 attention buffers alias into it.
    const size_t SEQ = (size_t)ML * DD;        // 417,792
    float* im2col = ws;
    float* embed  = ws + 3670016;
    float* qb  = ws;
    float* kb  = ws + 1 * SEQ;
    float* vb  = ws + 2 * SEQ;
    float* ob  = ws + 3 * SEQ;
    float* t1b = ws + 4 * SEQ;
    float* fub = ws + 5 * SEQ;

    size_t o = 3670016 + 393216;               // end of region0
    float* t_seq   = ws + o; o += SEQ;
    float* f_seq   = ws + o; o += SEQ;
    float* resb    = ws + o; o += SEQ;
    float* hb      = ws + o; o += SEQ;
    float* xzb     = ws + o; o += (size_t)ML * 2 * DI_;
    float* xxb     = ws + o; o += (size_t)ML * DI_;
    float* dblb    = ws + o; o += (size_t)ML * DBLP;
    float* dt_act  = ws + o; o += (size_t)ML * DI_;
    float* yb      = ws + o; o += (size_t)ML * DI_;
    float* featb   = ws + o; o += (size_t)BB * DD;
    float* h1b     = ws + o; o += (size_t)BB * 512;
    float* dtw_pt  = ws + o; o += (size_t)DEPTH_ * DI_ * 32;
    float* dtw_pf  = ws + o; o += (size_t)DEPTH_ * DI_ * 32;

    const int IMN = 2048 * 1792;
    const int PADN = DEPTH_ * DI_ * 32;

    // ---- pad dt_w (K=12 -> K=32) once per launch ----
    k_pad_dtw<<<(PADN + 255) / 256, 256, 0, stream>>>(t_dtw, dtw_pt);
    k_pad_dtw<<<(PADN + 255) / 256, 256, 0, stream>>>(f_dtw, dtw_pf);

    // ---- patch embeddings ----
    k_im2col_t<<<(IMN + 255) / 256, 256, 0, stream>>>(X, im2col);
    gemm(stream, im2col, 1792, t_pw, 1792, t_pb, embed, DD, 2048, DD, 1792, 0);
    k_assemble<<<(BB * LL * DD + 255) / 256, 256, 0, stream>>>(embed, t_tok, t_pos, t_seq);

    k_im2col_f<<<(IMN + 255) / 256, 256, 0, stream>>>(X, im2col);
    gemm(stream, im2col, 1792, f_pw, 1792, f_pb, embed, DD, 2048, DD, 1792, 0);
    k_assemble<<<(BB * LL * DD + 255) / 256, 256, 0, stream>>>(embed, f_tok, f_pos, f_seq);

    // ---- mamba backbones (12 layers each) ----
    run_backbone(stream, t_seq, resb, hb, xzb, xxb, dblb, dt_act, yb,
                 t_nw, t_inp, t_cw, t_cb, t_xp, dtw_pt, t_dtbv, t_alog, t_dd, t_op);
    run_backbone(stream, f_seq, resb, hb, xzb, xxb, dblb, dt_act, yb,
                 f_nw, f_inp, f_cw, f_cb, f_xp, dtw_pf, f_dtbv, f_alog, f_dd, f_op);

    // ---- cross attention fusion ----
    run_xattn(stream, t_seq, f_seq, att_t_w,  att_t_b,  qb, kb, vb, ob, t1b);
    run_xattn(stream, t1b,   f_seq, att_tf_w, att_tf_b, qb, kb, vb, ob, fub);
    k_feat<<<(BB * DD + 255) / 256, 256, 0, stream>>>(t_seq, fub, featb);

    // ---- heads: cls (128x10) then pos (128x3), concatenated in d_out ----
    gemm(stream, featb, DD, cls1_w, DD, cls1_b, h1b, 512, BB, 512, DD, 1 /*relu*/);
    gemm(stream, h1b, 512, cls2_w, 512, cls2_b, out, 10, BB, 10, 512, 0);
    gemm(stream, featb, DD, pos1_w, DD, pos1_b, h1b, 512, BB, 512, DD, 1 /*relu*/);
    gemm(stream, h1b, 512, pos2_w, 512, pos2_b, out + BB * 10, 3, BB, 3, 512, 0);
}
